// EvolvingFieldSystem_54494545052129
// MI455X (gfx1250) — matrix operand order repaired and checked by reading, not compiled, not run on hardware
//
#include <hip/hip_runtime.h>
#include <hip/hip_bf16.h>

typedef __attribute__((ext_vector_type(16))) _Float16 v16h;
typedef __attribute__((ext_vector_type(8)))  _Float16 v8h;
typedef __attribute__((ext_vector_type(8)))  float    v8f;

// ---------------------------------------------------------------------------
// Fragment helpers (CDNA5 wave32 WMMA f16 16x16x32 layouts, ISA 7.12.2)
// A (16xK): lane l -> row M=l&15; khalf=l>>4; element e:
//   K = ks*32 + (e<8 ? e + 8*khalf : e + 8 + 8*khalf)   (two 16B-contiguous runs)
// B (Kx16): lane l -> col N=l&15; K = same ks,khalf,e mapping as A (pre-packed per lane)
// C/D f32:  lane l, vgpr j -> D[j + 8*(l>>4)][l&15]
// ---------------------------------------------------------------------------
#define RS 72     // LDS activation row stride in halves (36 dwords: conflict-minimal)
#define TILES 8   // 16-point tiles per wave -> 1024 points per 256-thread block

#define WBAR() __builtin_amdgcn_wave_barrier()

__device__ inline v16h load_frag_a(const _Float16* row, int ks, int khalf) {
    v8h lo = *(const v8h*)(row + ks * 32 + 8 * khalf);
    v8h hi = *(const v8h*)(row + ks * 32 + 16 + 8 * khalf);
    return __builtin_shufflevector(lo, hi, 0, 1, 2, 3, 4, 5, 6, 7,
                                           8, 9, 10, 11, 12, 13, 14, 15);
}

__device__ inline v16h load_frag_w(const _Float16* p) {
    v8h lo = *(const v8h*)(p);
    v8h hi = *(const v8h*)(p + 8);
    return __builtin_shufflevector(lo, hi, 0, 1, 2, 3, 4, 5, 6, 7,
                                           8, 9, 10, 11, 12, 13, 14, 15);
}

__device__ inline v8f wmma_f16(v16h a, v16h b, v8f c) {
    return __builtin_amdgcn_wmma_f32_16x16x32_f16(false, a, false, b,
                                                  (short)0, c, false, false);
}

// single-instruction ReLU: v_med3_f32(x, 0, +inf)
__device__ inline float relu1(float x) {
    return __builtin_amdgcn_fmed3f(x, 0.f, __builtin_inff());
}

#define N_BLOBS 20
#define NFRAG   20   // 8 (fq_w1) + 8 (col_w1) + 2 (fq_w2 head) + 2 (col_w2 head)

// ---------------------------------------------------------------------------
// Kernel 1: sequential blob-ODE (one block), summary vector, and f16 weight
// fragments pre-packed in WMMA B-fragment order into d_ws.
// ws layout: [0,128)B  summary (32 f32)
//            [128,...) wfrag   (20 frags x 32 lanes x 16 halves = 20480 B)
// frag f: 0..7  = fq_w1  (ks*4+nt)      56->64
//         8..15 = col_w1 (8+ks*4+nt)    57->64
//         16..17= fq_w2 head (N=0 col)  64->1
//         18..19= col_w2 head (N=0..2)  64->3
// ---------------------------------------------------------------------------
__global__ __launch_bounds__(256) void prep_kernel(
    const float* __restrict__ initial_state,
    const float* __restrict__ dyn_w1, const float* __restrict__ dyn_b1,
    const float* __restrict__ dyn_w2, const float* __restrict__ dyn_b2,
    const float* __restrict__ coupling, const float* __restrict__ damping,
    const float* __restrict__ interaction,
    const float* __restrict__ summ_w, const float* __restrict__ summ_b,
    const float* __restrict__ fq_w1, const float* __restrict__ fq_w2,
    const float* __restrict__ col_w1, const float* __restrict__ col_w2,
    const int* __restrict__ tptr,
    float* __restrict__ ws_summary, _Float16* __restrict__ ws_frag)
{
    __shared__ float st[160];
    __shared__ float hh[64];
    __shared__ float dn[160];
    __shared__ float fo[N_BLOBS][3];

    const int tid = threadIdx.x;
    if (tid < 160) st[tid] = initial_state[tid];
    __syncthreads();

    // reproduce Python: n = int(t / 0.01); dt = float32(t / n)  (IEEE double)
    const double td = (double)tptr[0];
    int nsteps = (int)(td / 0.01);
    if (nsteps < 1) nsteps = 1;
    const float dt  = (float)(td / (double)nsteps);
    const float cpl = coupling[0];
    const float dmp = damping[0];

    for (int s = 0; s < nsteps; ++s) {
        if (tid < 64) {
            float acc = dyn_b1[tid];
            for (int k = 0; k < 160; ++k) acc += dyn_w1[tid * 160 + k] * st[k];
            hh[tid] = tanhf(acc);
        }
        __syncthreads();
        if (tid < 160) {
            float acc = dyn_b2[tid];
            for (int k = 0; k < 64; ++k) acc += dyn_w2[tid * 64 + k] * hh[k];
            dn[tid] = acc;
        } else if (tid < 160 + N_BLOBS) {
            const int i = tid - 160;
            const float xi = st[i * 8], yi = st[i * 8 + 1], zi = st[i * 8 + 2];
            float fx = 0.f, fy = 0.f, fz = 0.f;
            for (int j = 0; j < N_BLOBS; ++j) {
                const float dx = xi - st[j * 8];
                const float dy = yi - st[j * 8 + 1];
                const float dz = zi - st[j * 8 + 2];
                const float dist = sqrtf(dx * dx + dy * dy + dz * dz + 1e-6f);
                const float fm  = interaction[i * N_BLOBS + j] / (dist * dist + 1.0f);
                const float w   = fm / (dist + 1e-6f);
                fx += w * dx; fy += w * dy; fz += w * dz;
            }
            fo[i][0] = fx; fo[i][1] = fy; fo[i][2] = fz;
        }
        __syncthreads();
        float newv = 0.f;
        if (tid < 160) {
            const int i = tid >> 3, c = tid & 7;
            float ds;
            if (c < 3)       ds = st[i * 8 + 3 + c];
            else if (c < 6)  ds = -dmp * st[i * 8 + c] + cpl * fo[i][c - 3] + dn[i * 8 + c];
            else if (c == 6) ds = -0.05f * (st[i * 8 + 6] - 1.0f) + dn[i * 8 + 6];
            else             ds = -0.05f * (st[i * 8 + 7] - 0.5f) + dn[i * 8 + 7];
            newv = st[tid] + dt * ds;
        }
        __syncthreads();
        if (tid < 160) st[tid] = newv;
        __syncthreads();
    }

    if (tid < 32) {
        float acc = summ_b[tid];
        for (int k = 0; k < 160; ++k) acc += summ_w[tid * 160 + k] * st[k];
        ws_summary[tid] = acc;
    }

    // pack B fragments (f16): NFRAG*32 (frag,lane) pairs
    for (int q = tid; q < NFRAG * 32; q += 256) {
        const int f = q >> 5, l = q & 31;
        const int khalf = l >> 4, n16 = l & 15;
        for (int e = 0; e < 16; ++e) {
            float v = 0.f;
            const int kk = (e < 8) ? (8 * khalf + e) : (16 + 8 * khalf + (e - 8));
            if (f < 8) {
                const int ks = (f >> 2) & 1, K = ks * 32 + kk;
                const int n = (f & 3) * 16 + n16;
                if (K < 56) v = fq_w1[n * 56 + K];
            } else if (f < 16) {
                const int ks = (f >> 2) & 1, K = ks * 32 + kk;
                const int n = (f & 3) * 16 + n16;
                if (K < 57) v = col_w1[n * 57 + K];   // K==56 = density weight
            } else if (f < 18) {
                const int K = (f - 16) * 32 + kk;
                if (n16 == 0) v = fq_w2[K];
            } else {
                const int K = (f - 18) * 32 + kk;
                if (n16 < 3) v = col_w2[n16 * 64 + K];
            }
            ws_frag[(f * 32 + l) * 16 + e] = (_Float16)v;
        }
    }
}

// ---------------------------------------------------------------------------
// Kernel 2: per-point field MLP, fully on WMMA (20 wmma / 16-point tile).
// 256 thr = 8 waves; each wave does TILES tiles of 16 points.
// All activation staging is PER-WAVE LDS: no block barriers in the tile loop
// (wave-internal DS ops are architecturally in-order; wave_barrier only stops
// compiler reordering).
// ---------------------------------------------------------------------------
__global__ __launch_bounds__(256) void field_kernel(
    const float* __restrict__ p,
    const float* __restrict__ fq_b1, const float* __restrict__ fq_b2,
    const float* __restrict__ col_b1, const float* __restrict__ col_b2,
    const float* __restrict__ ws_summary, const _Float16* __restrict__ ws_frag,
    float* __restrict__ out, int N)
{
    __shared__ __align__(16) _Float16 wfrag[NFRAG * 32 * 16]; // 20 KB
    __shared__ __align__(16) _Float16 abuf[8][16 * RS];       // 18 KB per-wave acts
    __shared__ float ssum[32];
    __shared__ float sb_fq1[64], sb_col1[64], sb_misc[4];
    __shared__ float dens[8][16];
    __shared__ float colstg[8][16][3];

    const int tid = threadIdx.x;

    // coalesced copy of weight fragments into LDS (1280 uint4 total)
    {
        const uint4* src = (const uint4*)ws_frag;
        uint4* dst = (uint4*)wfrag;
#pragma unroll
        for (int i = 0; i < 5; ++i) dst[tid + 256 * i] = src[tid + 256 * i];
    }
    if (tid < 32) ssum[tid] = ws_summary[tid];
    else if (tid >= 64 && tid < 128) sb_fq1[tid - 64] = fq_b1[tid - 64];
    else if (tid >= 128 && tid < 192) sb_col1[tid - 128] = col_b1[tid - 128];
    else if (tid == 192) sb_misc[0] = fq_b2[0];
    else if (tid >= 193 && tid < 196) sb_misc[tid - 192] = col_b2[tid - 193];
    __syncthreads();   // the only block-wide barrier

    const int w = tid >> 5, lane = tid & 31;
    const int m = lane & 15, khalf = lane >> 4;
    _Float16* buf = &abuf[w][0];
    const _Float16* arow = buf + m * RS;
    _Float16* mrow = buf + m * RS;

    // pre-convert summary to f16 vectors (cols 24..55) + zero tail (56..63)
    v8h s16a, s16b, s16c, s16d, zero8 = {};
#pragma unroll
    for (int q = 0; q < 8; ++q) {
        s16a[q] = (_Float16)ssum[q];
        s16b[q] = (_Float16)ssum[8 + q];
        s16c[q] = (_Float16)ssum[16 + q];
        s16d[q] = (_Float16)ssum[24 + q];
    }

#pragma unroll 1
    for (int t = 0; t < TILES; ++t) {
        const long idx = (long)blockIdx.x * (TILES * 128) + t * 128 + w * 16 + m;
        const bool valid = idx < (long)N;

        // Phase A: lanes 0..15 write pos-enc (cols 0..23, 3x b128);
        //          lanes 16..31 write summary + zero tail (cols 24..63, 5x b128)
        if (lane < 16) {
            float c3[3] = {0.f, 0.f, 0.f};
            if (valid) { c3[0] = p[3 * idx]; c3[1] = p[3 * idx + 1]; c3[2] = p[3 * idx + 2]; }
            float pe[24];
#pragma unroll
            for (int i = 0; i < 4; ++i) {
                const float f = (float)(1 << i) * 6.28318530717958647692f;
#pragma unroll
                for (int d = 0; d < 3; ++d) {
                    float sv, cv;
                    __sincosf(f * c3[d], &sv, &cv);
                    pe[(2 * i + 0) * 3 + d] = sv;
                    pe[(2 * i + 1) * 3 + d] = cv;
                }
            }
            v8h e0, e1, e2;
#pragma unroll
            for (int q = 0; q < 8; ++q) {
                e0[q] = (_Float16)pe[q];
                e1[q] = (_Float16)pe[8 + q];
                e2[q] = (_Float16)pe[16 + q];
            }
            *(v8h*)(mrow)      = e0;
            *(v8h*)(mrow + 8)  = e1;
            *(v8h*)(mrow + 16) = e2;
        } else {
            *(v8h*)(mrow + 24) = s16a;
            *(v8h*)(mrow + 32) = s16b;
            *(v8h*)(mrow + 40) = s16c;
            *(v8h*)(mrow + 48) = s16d;
            *(v8h*)(mrow + 56) = zero8;
        }
        WBAR();

        // A fragments for field_in (reused for layer 2 with density patch)
        v16h a0 = load_frag_a(arow, 0, khalf);
        v16h a1 = load_frag_a(arow, 1, khalf);
        WBAR();

        // Layer 1: 56->64 + ReLU (8 wmma), hidden back to LDS as f16
#pragma unroll
        for (int nt = 0; nt < 4; ++nt) {
            const float bias = sb_fq1[nt * 16 + m];
            v8f c;
#pragma unroll
            for (int j = 0; j < 8; ++j) c[j] = bias;
            c = wmma_f16(a0, load_frag_w(&wfrag[((0 + nt) * 32 + lane) * 16]), c);
            c = wmma_f16(a1, load_frag_w(&wfrag[((4 + nt) * 32 + lane) * 16]), c);
#pragma unroll
            for (int j = 0; j < 8; ++j)
                buf[(j + 8 * khalf) * RS + nt * 16 + m] = (_Float16)relu1(c[j]);
        }
        WBAR();

        // Density head: h(16x64) x fq_w2 via 2 wmma (column N=0)
        {
            v16h h0 = load_frag_a(arow, 0, khalf);
            v16h h1 = load_frag_a(arow, 1, khalf);
            const float bias = (m == 0) ? sb_misc[0] : 0.f;
            v8f c;
#pragma unroll
            for (int j = 0; j < 8; ++j) c[j] = bias;
            c = wmma_f16(h0, load_frag_w(&wfrag[(16 * 32 + lane) * 16]), c);
            c = wmma_f16(h1, load_frag_w(&wfrag[(17 * 32 + lane) * 16]), c);
            if (m == 0) {
#pragma unroll
                for (int j = 0; j < 8; ++j) dens[w][j + 8 * khalf] = c[j];
            }
        }
        WBAR();

        // softplus + density output; write activated value back for the patch
        if (lane < 16) {
            const float x = dens[w][m];
            const float d = fmaxf(x, 0.f) + log1pf(__expf(-fabsf(x)));
            dens[w][m] = d;
            if (valid) out[idx] = d;
        }
        WBAR();

        // Patch density into A fragment at K=56 (ks=1, khalf=1, e=8)
        if (khalf == 1) a1[8] = (_Float16)dens[w][m];
        WBAR();

        // Layer 2: 57->64 + ReLU (8 wmma), hidden back to LDS as f16
#pragma unroll
        for (int nt = 0; nt < 4; ++nt) {
            const float bias = sb_col1[nt * 16 + m];
            v8f c;
#pragma unroll
            for (int j = 0; j < 8; ++j) c[j] = bias;
            c = wmma_f16(a0, load_frag_w(&wfrag[((8 + nt) * 32 + lane) * 16]), c);
            c = wmma_f16(a1, load_frag_w(&wfrag[((12 + nt) * 32 + lane) * 16]), c);
#pragma unroll
            for (int j = 0; j < 8; ++j)
                buf[(j + 8 * khalf) * RS + nt * 16 + m] = (_Float16)relu1(c[j]);
        }
        WBAR();

        // Color head: hc(16x64) x col_w2 via 2 wmma (columns N=0..2)
        {
            v16h g0 = load_frag_a(arow, 0, khalf);
            v16h g1 = load_frag_a(arow, 1, khalf);
            const float bias = (m < 3) ? sb_misc[1 + m] : 0.f;
            v8f c;
#pragma unroll
            for (int j = 0; j < 8; ++j) c[j] = bias;
            c = wmma_f16(g0, load_frag_w(&wfrag[(18 * 32 + lane) * 16]), c);
            c = wmma_f16(g1, load_frag_w(&wfrag[(19 * 32 + lane) * 16]), c);
            if (m < 3) {
#pragma unroll
                for (int j = 0; j < 8; ++j) colstg[w][j + 8 * khalf][m] = c[j];
            }
        }
        WBAR();

        // sigmoid + color output (single b96 store per point)
        if (lane < 16 && valid) {
            float3 col;
            col.x = 1.f / (1.f + __expf(-colstg[w][m][0]));
            col.y = 1.f / (1.f + __expf(-colstg[w][m][1]));
            col.z = 1.f / (1.f + __expf(-colstg[w][m][2]));
            *(float3*)(out + (long)N + 3 * idx) = col;
        }
        WBAR();
    }
}

extern "C" void kernel_launch(void* const* d_in, const int* in_sizes, int n_in,
                              void* d_out, int out_size, void* d_ws, size_t ws_size,
                              hipStream_t stream) {
    const float* p             = (const float*)d_in[0];
    const float* initial_state = (const float*)d_in[1];
    const float* dyn_w1        = (const float*)d_in[2];
    const float* dyn_b1        = (const float*)d_in[3];
    const float* dyn_w2        = (const float*)d_in[4];
    const float* dyn_b2        = (const float*)d_in[5];
    const float* coupling      = (const float*)d_in[6];
    const float* damping       = (const float*)d_in[7];
    const float* interaction   = (const float*)d_in[8];
    const float* summ_w        = (const float*)d_in[9];
    const float* summ_b        = (const float*)d_in[10];
    const float* fq_w1         = (const float*)d_in[11];
    const float* fq_b1         = (const float*)d_in[12];
    const float* fq_w2         = (const float*)d_in[13];
    const float* fq_b2         = (const float*)d_in[14];
    const float* col_w1        = (const float*)d_in[15];
    const float* col_b1        = (const float*)d_in[16];
    const float* col_w2        = (const float*)d_in[17];
    const float* col_b2        = (const float*)d_in[18];
    const int*   tptr          = (const int*)d_in[19];

    float*     ws_summary = (float*)d_ws;
    _Float16*  ws_frag    = (_Float16*)((char*)d_ws + 128);

    const int N = in_sizes[0] / 3;

    prep_kernel<<<1, 256, 0, stream>>>(initial_state, dyn_w1, dyn_b1, dyn_w2,
                                       dyn_b2, coupling, damping, interaction,
                                       summ_w, summ_b, fq_w1, fq_w2, col_w1,
                                       col_w2, tptr, ws_summary, ws_frag);

    const int ptsPerBlock = TILES * 128;
    const int nblocks = (N + ptsPerBlock - 1) / ptsPerBlock;
    field_kernel<<<nblocks, 256, 0, stream>>>(p, fq_b1, fq_b2, col_b1, col_b2,
                                              ws_summary, ws_frag,
                                              (float*)d_out, N);
}
